// NET_49374944035367
// MI455X (gfx1250) — compile-verified, matrix operand order checked by reading, not run on hardware
//
#include <hip/hip_runtime.h>
#include <hip/hip_bf16.h>

// ---------------------------------------------------------------------------
// Model dims
// ---------------------------------------------------------------------------
#define BATCH 32
#define TSEQ  719
#define VOCAB 6000
#define DMODEL 256
#define GATES 1024           // 4*D
#define MROWS (BATCH*TSEQ)   // 23008, row m = b*TSEQ + t  -> coalesced output
#define MPAD  23040          // 360 * 64
#define NPAD_DEC 6016        // 94 * 64
#define EPSV 1e-5f

typedef __bf16 bf16_t;
typedef __attribute__((ext_vector_type(16))) __bf16 v16bf;
typedef __attribute__((ext_vector_type(8)))  __bf16 v8bf;
typedef __attribute__((ext_vector_type(8)))  float  v8f;
typedef __attribute__((ext_vector_type(2)))  float  v2f;
typedef __attribute__((ext_vector_type(4)))  unsigned int u32x4;
typedef __attribute__((ext_vector_type(8)))  int    i32x8;
typedef __attribute__((ext_vector_type(4)))  int    i32x4;

#define USE_TDM 1

__device__ __forceinline__ float sigm(float x) { return 1.f / (1.f + __expf(-x)); }

__device__ __forceinline__ v16bf join8(v8bf lo, v8bf hi) {
    return __builtin_shufflevector(lo, hi, 0,1,2,3,4,5,6,7,8,9,10,11,12,13,14,15);
}

// ---------------------------------------------------------------------------
// Kernel 1: CNN stack (per-sample instance-norm conv chain). One block/board.
// Tiny workload; plain VALU. Output: a_buf[32][361] f32.
// ---------------------------------------------------------------------------
__global__ void __launch_bounds__(512)
conv_stack_kernel(const float* __restrict__ board, const float* __restrict__ ck,
                  const float* __restrict__ cbv, const float* __restrict__ gam,
                  const float* __restrict__ bet, float* __restrict__ a_buf) {
    const int b = blockIdx.x, tid = threadIdx.x;
    __shared__ float A[361], R[361], red[512];
    __shared__ float k9[9], s_cb, s_g, s_bt;
    if (tid < 9)  k9[tid] = ck[tid];
    if (tid == 9)  s_cb = cbv[0];
    if (tid == 10) s_g  = gam[0];
    if (tid == 11) s_bt = bet[0];
    const bool act = tid < 361;
    const int y = tid / 19, x = tid % 19;
    if (act) A[tid] = board[b*361 + tid];
    __syncthreads();

    auto conv_bn = [&](bool relu_out) {
        float s = 0.f;
        if (act) {
            #pragma unroll
            for (int dy = -1; dy <= 1; ++dy)
                #pragma unroll
                for (int dx = -1; dx <= 1; ++dx) {
                    int yy = y + dy, xx = x + dx;
                    if (yy >= 0 && yy < 19 && xx >= 0 && xx < 19)
                        s += A[yy*19 + xx] * k9[(dy+1)*3 + (dx+1)];
                }
            s += s_cb;
        }
        red[tid] = act ? s : 0.f; __syncthreads();
        for (int st = 256; st > 0; st >>= 1) { if (tid < st) red[tid] += red[tid + st]; __syncthreads(); }
        float m = red[0] * (1.f/361.f);
        __syncthreads();
        float d = act ? (s - m) : 0.f;
        red[tid] = d * d; __syncthreads();
        for (int st = 256; st > 0; st >>= 1) { if (tid < st) red[tid] += red[tid + st]; __syncthreads(); }
        float v = red[0] * (1.f/361.f);
        float o = (s - m) * rsqrtf(v + EPSV) * s_g + s_bt;
        __syncthreads();
        if (act) A[tid] = relu_out ? fmaxf(o, 0.f) : o;
        __syncthreads();
    };

    conv_bn(true);
    if (act) R[tid] = A[tid];
    __syncthreads();
    #pragma unroll 1
    for (int it = 0; it < 3; ++it) {
        conv_bn(true);
        conv_bn(false);
        if (act) { float v = fmaxf(A[tid] + R[tid], 0.f); A[tid] = v; R[tid] = v; }
        __syncthreads();
    }
    conv_bn(true);
    if (act) a_buf[b*361 + tid] = A[tid];
}

// ---------------------------------------------------------------------------
// Kernel 2: feat = a_buf @ lin_W^T + lin_b  -> row t=0 of X (bf16)
// ---------------------------------------------------------------------------
__global__ void __launch_bounds__(256)
feat_kernel(const float* __restrict__ a_buf, const float* __restrict__ lin_W,
            const float* __restrict__ lin_b, bf16_t* __restrict__ X) {
    const int b = blockIdx.x, d = threadIdx.x;
    __shared__ float av[361];
    for (int p = d; p < 361; p += 256) av[p] = a_buf[b*361 + p];
    __syncthreads();
    float s = lin_b[d];
    const float* w = lin_W + (size_t)d * 361;
    for (int p = 0; p < 361; ++p) s += av[p] * w[p];
    X[((size_t)b * TSEQ) * DMODEL + d] = (bf16_t)s;   // row m = b*T + 0
}

// ---------------------------------------------------------------------------
// Kernel 3: embedding gather (masked) -> X rows t=1..718 (bf16)
// ---------------------------------------------------------------------------
__global__ void __launch_bounds__(256)
embed_kernel(const int* __restrict__ comment, const float* __restrict__ emb,
             bf16_t* __restrict__ X) {
    const int row = blockIdx.x;             // 0 .. 32*718-1
    const int b = row / (TSEQ - 1), t = row % (TSEQ - 1) + 1;
    int tok = comment[b * TSEQ + t];
    const int d = threadIdx.x;
    float v = 0.f;
    if (tok >= 0) {
        tok = tok >= VOCAB ? VOCAB - 1 : tok;
        v = emb[(size_t)tok * DMODEL + d];
    }
    X[((size_t)(b * TSEQ + t)) * DMODEL + d] = (bf16_t)v;
}

// ---------------------------------------------------------------------------
// Kernel 4a/4b/4c: weight repacks into WMMA-fragment-linear layouts.
//   bf16 B packed as [g=K/8][Npad][8] (16B per lane fragment-load)
//   Whh packed f32 pairs for V_WMMA_F32_16X16X4_F32 B-fragments
// ---------------------------------------------------------------------------
__global__ void pack_b_bf16(const float* __restrict__ W, bf16_t* __restrict__ Bp,
                            int Nreal, int Npad, int total) {
    int i = blockIdx.x * 256 + threadIdx.x;         // i = g*Npad + n
    if (i >= total) return;
    int g = i / Npad, n = i % Npad;
    #pragma unroll
    for (int j = 0; j < 8; ++j)
        Bp[(size_t)i * 8 + j] = (n < Nreal) ? (bf16_t)W[(size_t)n * DMODEL + g*8 + j]
                                            : (bf16_t)0.f;
}

__global__ void pack_whh(const float* __restrict__ Whh, float* __restrict__ WhhP) {
    int i = blockIdx.x * 256 + threadIdx.x;         // i = kk2*1024 + n, kk2 in [0,128)
    if (i >= 128 * GATES) return;
    int kk2 = i >> 10, n = i & 1023;
    int k = (kk2 >> 1) * 4 + (kk2 & 1) * 2;         // lane-half K pair base
    WhhP[2*i]   = Whh[(size_t)n * DMODEL + k];
    WhhP[2*i+1] = Whh[(size_t)n * DMODEL + k + 1];
}

// ---------------------------------------------------------------------------
// Kernel 5: Xpre = X @ Wih^T + (bih+bhh)   (bf16 WMMA, f32 accumulate)
// block tile 64x64, 8 waves, wave tile 16x32 (2 accumulators)
// ---------------------------------------------------------------------------
__global__ void __launch_bounds__(256)
xpre_gemm_kernel(const bf16_t* __restrict__ X, const bf16_t* __restrict__ Bp,
                 const float* __restrict__ bih, const float* __restrict__ bhh,
                 float* __restrict__ Xpre) {
    const int lane = threadIdx.x & 31, wave = threadIdx.x >> 5;
    const int l15 = lane & 15, hi = lane >> 4;
    const int m0 = blockIdx.x * 64 + (wave & 3) * 16;
    const int n0 = blockIdx.y * 64 + (wave >> 2) * 32;
    v8f acc0 = {0,0,0,0,0,0,0,0}, acc1 = acc0;
    const bf16_t* arow = X + (size_t)(m0 + l15) * DMODEL;
    #pragma unroll
    for (int kk = 0; kk < 8; ++kk) {
        int kbase = kk * 32 + hi * 8;
        v16bf af = join8(*(const v8bf*)(arow + kbase), *(const v8bf*)(arow + kbase + 16));
        int glo = kk * 4 + hi, ghi = glo + 2;
        v16bf b0 = join8(*(const v8bf*)(Bp + ((size_t)glo * GATES + n0 + l15) * 8),
                         *(const v8bf*)(Bp + ((size_t)ghi * GATES + n0 + l15) * 8));
        v16bf b1 = join8(*(const v8bf*)(Bp + ((size_t)glo * GATES + n0 + 16 + l15) * 8),
                         *(const v8bf*)(Bp + ((size_t)ghi * GATES + n0 + 16 + l15) * 8));
        acc0 = __builtin_amdgcn_wmma_f32_16x16x32_bf16(false, af, false, b0, (short)0, acc0, false, false);
        acc1 = __builtin_amdgcn_wmma_f32_16x16x32_bf16(false, af, false, b1, (short)0, acc1, false, false);
    }
    #pragma unroll
    for (int r = 0; r < 8; ++r) {
        int m = m0 + r + hi * 8;
        int na = n0 + l15, nb = n0 + 16 + l15;
        Xpre[(size_t)m * GATES + na] = acc0[r] + bih[na] + bhh[na];
        Xpre[(size_t)m * GATES + nb] = acc1[r] + bih[nb] + bhh[nb];
    }
}

// ---------------------------------------------------------------------------
// Kernel 6: sequential LSTM recurrence. ONE workgroup, 512 threads (16 waves).
// Gates via V_WMMA_F32_16X16X4_F32 (full f32 precision in the recurrence).
// Wave w owns hidden-dim slice d=[16w,16w+16) across all 4 gates, so the
// i/f/g/o tiles for a given (b,d) live in the SAME lane -> c/h update fully
// in registers. h is broadcast through LDS; c persists in VGPRs all 719 steps.
// ---------------------------------------------------------------------------
__global__ void __launch_bounds__(512)
lstm_kernel(const float* __restrict__ Xpre, const float* __restrict__ WhhP,
            bf16_t* __restrict__ H) {
    const int tid = threadIdx.x, lane = tid & 31, wave = tid >> 5;
    const int l15 = lane & 15, hi = lane >> 4;
    const int dcol = wave * 16 + l15;
    __shared__ float h_lds[BATCH * DMODEL];     // 32 KB
    for (int i = tid; i < BATCH * DMODEL; i += 512) h_lds[i] = 0.f;
    __syncthreads();
    v8f c0 = {0,0,0,0,0,0,0,0}, c1 = c0;

    for (int t = 0; t < TSEQ; ++t) {
        v8f acc[2][4];
        #pragma unroll
        for (int mt = 0; mt < 2; ++mt)
            #pragma unroll
            for (int g = 0; g < 4; ++g)
                #pragma unroll
                for (int r = 0; r < 8; ++r) {
                    int b = mt * 16 + r + hi * 8;
                    acc[mt][g][r] = Xpre[((size_t)(b * TSEQ + t)) * GATES + g * 256 + dcol];
                }
        // h @ Whh^T : K=256 as 64 x (16x16x4 f32 WMMA)
        #pragma unroll 4
        for (int kk = 0; kk < 64; ++kk) {
            int k0 = kk * 4 + hi * 2;
            v2f a0 = *(const v2f*)&h_lds[l15 * DMODEL + k0];
            v2f a1 = *(const v2f*)&h_lds[(16 + l15) * DMODEL + k0];
            int bbase = (kk * 2 + hi) * GATES + dcol;
            #pragma unroll
            for (int g = 0; g < 4; ++g) {
                v2f bf = *(const v2f*)&WhhP[(size_t)(bbase + g * 256) * 2];
                acc[0][g] = __builtin_amdgcn_wmma_f32_16x16x4_f32(false, a0, false, bf, (short)0, acc[0][g], false, false);
                acc[1][g] = __builtin_amdgcn_wmma_f32_16x16x4_f32(false, a1, false, bf, (short)0, acc[1][g], false, false);
            }
        }
        __syncthreads();                         // all h_lds reads complete
        #pragma unroll
        for (int mt = 0; mt < 2; ++mt) {
            v8f& cc = mt ? c1 : c0;
            #pragma unroll
            for (int r = 0; r < 8; ++r) {
                float ig = sigm(acc[mt][0][r]);
                float fg = sigm(acc[mt][1][r]);
                float gg = tanhf(acc[mt][2][r]);
                float og = sigm(acc[mt][3][r]);
                float cv = fg * cc[r] + ig * gg;
                cc[r] = cv;
                float hv = og * tanhf(cv);
                int b = mt * 16 + r + hi * 8;
                h_lds[b * DMODEL + dcol] = hv;
                H[((size_t)(b * TSEQ + t)) * DMODEL + dcol] = (bf16_t)hv;
            }
        }
        __syncthreads();                         // h published for next step
    }
}

// ---------------------------------------------------------------------------
// Kernel 7: decoder GEMM  out = H @ dec_W^T + dec_b  (bf16 WMMA, f32 acc)
// Rows m = b*T + t  ->  out[m*V + n] is the exact (B,T,V) layout: contiguous
// coalesced stores of the 552 MB output (the HBM-bound part, ~24us floor).
// B slab staged into LDS via the Tensor Data Mover; cluster barrier shown
// (NOP outside cluster dispatch per ISA).
// ---------------------------------------------------------------------------
__global__ void __launch_bounds__(256)
dec_gemm_kernel(const bf16_t* __restrict__ H, const bf16_t* __restrict__ Bp,
                const float* __restrict__ dec_b, float* __restrict__ out) {
    const int lane = threadIdx.x & 31, wave = threadIdx.x >> 5;
    const int l15 = lane & 15, hi = lane >> 4;
    const int m0 = blockIdx.x * 64 + (wave & 3) * 16;
    const int nblk = blockIdx.y * 64;
    const int n0 = nblk + (wave >> 2) * 32;

    int cid = __builtin_amdgcn_cluster_id_x();
    if (cid > 15) return;    // impossible (4-bit id); keeps intrinsic live

#if USE_TDM
    // Stage this block's B slab (32 k-groups x 64 cols x 16B = 32 KB) via TDM.
    __shared__ __align__(128) unsigned char bslab[32 * 64 * 16];
    if (wave == 0) {
        unsigned long long ga = (unsigned long long)(const void*)Bp
                              + (unsigned long long)nblk * 16ull;
        unsigned lds = (unsigned)(unsigned long long)(void*)&bslab[0];
        const unsigned rowb = NPAD_DEC * 16u;          // bytes per k-group row
        u32x4 g0; g0[0] = 1u;                          // count=1 user descriptor
        g0[1] = lds;
        g0[2] = (unsigned)(ga & 0xFFFFFFFFu);
        g0[3] = (unsigned)((ga >> 32) & 0x1FFFFFFu) | (2u << 30);   // type=2
        i32x8 g1 = {0,0,0,0,0,0,0,0};
        g1[1] = (int)((rowb & 0xFFFFu) << 16);          // tensor_dim0[15:0]
        g1[2] = (int)((rowb >> 16) & 0xFFFFu) | (int)(32u << 16); // dim0 hi | dim1
        g1[3] = (int)(1024u << 16);                     // tile_dim0 = 1024 B
        g1[4] = 32;                                     // tile_dim1 = 32 rows
        g1[5] = (int)rowb;                              // dim0_stride lo
        i32x4 gz = {0,0,0,0};
#if defined(__clang_major__) && __clang_major__ >= 23
        i32x8 g4 = {0,0,0,0,0,0,0,0};
        __builtin_amdgcn_tensor_load_to_lds(g0, g1, gz, gz, g4, 0);
#else
        __builtin_amdgcn_tensor_load_to_lds(g0, g1, gz, gz, 0);
#endif
        __builtin_amdgcn_s_wait_tensorcnt(0);
    }
    __syncthreads();
    __builtin_amdgcn_s_cluster_barrier();   // NOP unless dispatched as cluster
#endif

    v8f acc0 = {0,0,0,0,0,0,0,0}, acc1 = acc0;
    const bf16_t* arow = H + (size_t)(m0 + l15) * DMODEL;
    __builtin_prefetch(arow + 64 * DMODEL, 0, 1);       // warm next M-block
    const int ca = (n0 - nblk) + l15, cb = (n0 - nblk) + 16 + l15;
    #pragma unroll
    for (int kk = 0; kk < 8; ++kk) {
        int kbase = kk * 32 + hi * 8;
        v16bf af = join8(*(const v8bf*)(arow + kbase), *(const v8bf*)(arow + kbase + 16));
        int glo = kk * 4 + hi, ghi = glo + 2;
#if USE_TDM
        v16bf b0 = join8(*(const v8bf*)&bslab[(glo * 64 + ca) * 16],
                         *(const v8bf*)&bslab[(ghi * 64 + ca) * 16]);
        v16bf b1 = join8(*(const v8bf*)&bslab[(glo * 64 + cb) * 16],
                         *(const v8bf*)&bslab[(ghi * 64 + cb) * 16]);
#else
        v16bf b0 = join8(*(const v8bf*)(Bp + ((size_t)glo * NPAD_DEC + n0 + l15) * 8),
                         *(const v8bf*)(Bp + ((size_t)ghi * NPAD_DEC + n0 + l15) * 8));
        v16bf b1 = join8(*(const v8bf*)(Bp + ((size_t)glo * NPAD_DEC + n0 + 16 + l15) * 8),
                         *(const v8bf*)(Bp + ((size_t)ghi * NPAD_DEC + n0 + 16 + l15) * 8));
#endif
        acc0 = __builtin_amdgcn_wmma_f32_16x16x32_bf16(false, af, false, b0, (short)0, acc0, false, false);
        acc1 = __builtin_amdgcn_wmma_f32_16x16x32_bf16(false, af, false, b1, (short)0, acc1, false, false);
    }
    #pragma unroll
    for (int r = 0; r < 8; ++r) {
        int m = m0 + r + hi * 8;
        if (m < MROWS) {
            int na = n0 + l15, nb = n0 + 16 + l15;
            size_t base = (size_t)m * VOCAB;
            if (na < VOCAB) out[base + na] = acc0[r] + dec_b[na];
            if (nb < VOCAB) out[base + nb] = acc1[r] + dec_b[nb];
        }
    }
}

// ---------------------------------------------------------------------------
// Host: workspace layout + launches (all on `stream`, graph-capture safe).
// Workspace need: ~123 MB (Xpre dominates at 94 MB).
// ---------------------------------------------------------------------------
static inline size_t alup(size_t x) { return (x + 255) & ~(size_t)255; }

extern "C" void kernel_launch(void* const* d_in, const int* in_sizes, int n_in,
                              void* d_out, int out_size, void* d_ws, size_t ws_size,
                              hipStream_t stream) {
    const float* board   = (const float*)d_in[0];
    const int*   comment = (const int*)  d_in[1];
    const float* conv_k  = (const float*)d_in[2];
    const float* conv_b  = (const float*)d_in[3];
    const float* bn_g    = (const float*)d_in[4];
    const float* bn_b    = (const float*)d_in[5];
    const float* emb     = (const float*)d_in[6];
    const float* lin_W   = (const float*)d_in[7];
    const float* lin_b   = (const float*)d_in[8];
    const float* Wih     = (const float*)d_in[9];
    const float* Whh     = (const float*)d_in[10];
    const float* bih     = (const float*)d_in[11];
    const float* bhh     = (const float*)d_in[12];
    const float* dec_W   = (const float*)d_in[13];
    const float* dec_b   = (const float*)d_in[14];
    float* out = (float*)d_out;

    char* w = (char*)d_ws; size_t off = 0;
    float*  a_buf = (float*) (w + off); off = alup(off + (size_t)BATCH*361*4);
    bf16_t* X     = (bf16_t*)(w + off); off = alup(off + (size_t)MPAD*DMODEL*2);
    bf16_t* WihP  = (bf16_t*)(w + off); off = alup(off + (size_t)32*GATES*8*2);
    float*  WhhP  = (float*) (w + off); off = alup(off + (size_t)128*GATES*2*4);
    bf16_t* decP  = (bf16_t*)(w + off); off = alup(off + (size_t)32*NPAD_DEC*8*2);
    float*  Xpre  = (float*) (w + off); off = alup(off + (size_t)MPAD*GATES*4);
    bf16_t* H     = (bf16_t*)(w + off); off = alup(off + (size_t)MPAD*DMODEL*2);
    (void)ws_size; (void)in_sizes; (void)n_in; (void)out_size;

    conv_stack_kernel<<<BATCH, 512, 0, stream>>>(board, conv_k, conv_b, bn_g, bn_b, a_buf);
    feat_kernel<<<BATCH, 256, 0, stream>>>(a_buf, lin_W, lin_b, X);
    embed_kernel<<<BATCH*(TSEQ-1), 256, 0, stream>>>(comment, emb, X);

    pack_b_bf16<<<(32*GATES + 255)/256, 256, 0, stream>>>(Wih, WihP, GATES, GATES, 32*GATES);
    pack_whh<<<(128*GATES + 255)/256, 256, 0, stream>>>(Whh, WhhP);
    pack_b_bf16<<<(32*NPAD_DEC + 255)/256, 256, 0, stream>>>(dec_W, decP, VOCAB, NPAD_DEC, 32*NPAD_DEC);

    xpre_gemm_kernel<<<dim3(MPAD/64, GATES/64), 256, 0, stream>>>(X, WihP, bih, bhh, Xpre);
    lstm_kernel<<<1, 512, 0, stream>>>(Xpre, WhhP, H);
    dec_gemm_kernel<<<dim3(MPAD/64, NPAD_DEC/64), 256, 0, stream>>>(H, decP, dec_b, out);
}
